// DualAttentionBlock_80547816669483
// MI455X (gfx1250) — matrix operand order repaired and checked by reading, not compiled, not run on hardware
//
#include <hip/hip_runtime.h>
#include <hip/hip_bf16.h>
#include <math.h>

// ---------------- types / WMMA helpers (gfx1250 wave32) ----------------
typedef __attribute__((ext_vector_type(16))) __bf16 v16bf;
typedef __attribute__((ext_vector_type(8)))  float  v8f;

__device__ __forceinline__ v8f vzero() {
  v8f z;
#pragma unroll
  for (int i = 0; i < 8; ++i) z[i] = 0.f;
  return z;
}

// A-matrix 16x32 bf16 fragment (ISA 7.12.2): lane r (r=lane&15) holds row r.
// VGPR j<4: K = 2j + 8*hi ; VGPR j>=4: K = 2j+8 + 8*hi   (hi = lane>>4)
__device__ __forceinline__ v16bf load_a16(const __bf16* base, int stride, int lane) {
  int r = lane & 15, hi = lane >> 4;
  const __bf16* p = base + (size_t)r * stride;
  v16bf f;
#pragma unroll
  for (int j = 0; j < 8; ++j) {
    int k = ((j < 4) ? (2 * j) : (2 * j + 8)) + 8 * hi;
    f[2 * j]     = p[k];
    f[2 * j + 1] = p[k + 1];
  }
  return f;
}

// B-matrix 32x16 bf16 fragment, loaded from B^T stored row-major [N][K]:
// lane r holds column r of B (= row r of B^T); same K layout as A: VGPR j<4: K=2j+8*hi, j>=4: K=2j+8+8*hi
__device__ __forceinline__ v16bf load_b16(const __bf16* baseT, int stride, int lane) {
  int r = lane & 15, hi = lane >> 4;
  const __bf16* p = baseT + (size_t)r * stride;
  v16bf f;
#pragma unroll
  for (int j = 0; j < 8; ++j) {
    int k = ((j < 4) ? (2 * j) : (2 * j + 8)) + 8 * hi;
    f[2 * j]     = p[k];
    f[2 * j + 1] = p[k + 1];
  }
  return f;
}

__device__ __forceinline__ v8f wmma_bf16(v16bf a, v16bf b, v8f c) {
  return __builtin_amdgcn_wmma_f32_16x16x32_bf16(false, a, false, b, (short)0, c, false, false);
}

// C/D 16x16 f32 tile: element e of lane l -> row = e + 8*(l>>4), col = l&15

// ---------------- constants ----------------
#define DIMC   180
#define HEADS  6
#define HD     30
#define WIN    16
#define NW     256              // tokens per window
#define LTOT   4096             // 64*64
#define MROWS  32768            // B*L
#define SCALE_F 0.18257418583505536f // 30^-0.5

// ---------------- weight convert: dst[Npad][Kpad] = src[K][N]^T (bf16, zero pad) ----
__global__ void cvt_wt_kernel(const float* __restrict__ src, __bf16* __restrict__ dst,
                              int K, int N, int Kpad, int Npad) {
  int i = blockIdx.x * 256 + threadIdx.x;
  if (i >= Npad * Kpad) return;
  int n = i / Kpad, k = i % Kpad;
  float v = (k < K && n < N) ? src[(size_t)k * N + n] : 0.f;
  dst[i] = (__bf16)v;
}

// ---------------- dynamic position bias MLP (961 rows) ----------------
__device__ void ln45_relu(const float* in, const float* g, const float* b, float* out) {
  float s = 0.f, sq = 0.f;
  for (int j = 0; j < 45; ++j) { s += in[j]; sq += in[j] * in[j]; }
  float mean = s * (1.f / 45.f);
  float var  = sq * (1.f / 45.f) - mean * mean;
  float rstd = rsqrtf(var + 1e-5f);
  for (int j = 0; j < 45; ++j) out[j] = fmaxf((in[j] - mean) * rstd * g[j] + b[j], 0.f);
}

__global__ void posbias_kernel(const float* __restrict__ ppw, const float* __restrict__ ppb,
                               const float* g1, const float* b1, const float* w1, const float* bb1,
                               const float* g2, const float* b2, const float* w2, const float* bb2,
                               const float* g3, const float* b3, const float* w3, const float* bb3,
                               float* __restrict__ posT) {
  int i = blockIdx.x * 256 + threadIdx.x;
  if (i >= 961) return;
  float c0 = (float)(i / 31 - 15), c1 = (float)(i % 31 - 15);
  float xa[45], xb[45], t1[45];
  for (int j = 0; j < 45; ++j) xa[j] = c0 * ppw[j] + c1 * ppw[45 + j] + ppb[j];
  ln45_relu(xa, g1, b1, t1);
  for (int k = 0; k < 45; ++k) {
    float a = bb1[k];
    for (int j = 0; j < 45; ++j) a += t1[j] * w1[j * 45 + k];
    xb[k] = a;
  }
  ln45_relu(xb, g2, b2, t1);
  for (int k = 0; k < 45; ++k) {
    float a = bb2[k];
    for (int j = 0; j < 45; ++j) a += t1[j] * w2[j * 45 + k];
    xa[k] = a;
  }
  ln45_relu(xa, g3, b3, t1);
  for (int h = 0; h < 6; ++h) {
    float a = bb3[h];
    for (int j = 0; j < 45; ++j) a += t1[j] * w3[j * 6 + h];
    posT[i * 6 + h] = a;
  }
}

// ---------------- LayerNorm: fp32 [M][180] -> bf16 [M][192] (zero padded) --------
__global__ __launch_bounds__(256) void ln_kernel(const float* __restrict__ x,
                                                 const float* __restrict__ g,
                                                 const float* __restrict__ b,
                                                 __bf16* __restrict__ out) {
  int wv = threadIdx.x >> 5, lane = threadIdx.x & 31;
  int m = blockIdx.x * 8 + wv;
  const float* row = x + (size_t)m * DIMC;
  float v[6], s = 0.f, sq = 0.f;
#pragma unroll
  for (int j = 0; j < 6; ++j) {
    int idx = lane + 32 * j;
    float val = (idx < DIMC) ? row[idx] : 0.f;
    v[j] = val; s += val; sq += val * val;
  }
#pragma unroll
  for (int d = 16; d > 0; d >>= 1) { s += __shfl_xor(s, d, 32); sq += __shfl_xor(sq, d, 32); }
  float mean = s * (1.f / 180.f);
  float var  = sq * (1.f / 180.f) - mean * mean;
  float rstd = rsqrtf(var + 1e-5f);
#pragma unroll
  for (int j = 0; j < 6; ++j) {
    int idx = lane + 32 * j;
    float y = (idx < DIMC) ? ((v[j] - mean) * rstd * g[idx] + b[idx]) : 0.f;
    out[(size_t)m * 192 + idx] = (__bf16)y;
  }
}

// ---------------- generic bf16 WMMA GEMM: C = A[M][Kpad] @ Wt^T + bias ----------
// Wt stored transposed+padded [Npad][Kpad]. Block = 8 waves, 128 rows x 16 cols.
// EPI 0: store bf16 (cols<180 scaled by qscale)     [QKV]
// EPI 1: +bias, +resid fp32, store fp32 (col<Ncols) [proj]
// EPI 2: +bias, exact GELU, store bf16              [fc1]
// EPI 3: +bias, +resid fp32, store fp32             [fc2 -> d_out]
template <int EPI>
__global__ __launch_bounds__(256) void gemm_bf16_kernel(
    const __bf16* __restrict__ A, int lda,
    const __bf16* __restrict__ Wt,
    const float* __restrict__ bias, int Ncols, int Kpad,
    void* __restrict__ Cout, int ldc,
    const float* __restrict__ resid, float qscale) {
  __shared__ __bf16 Ws[16 * 720];
  int n0 = blockIdx.y * 16;
  for (int i = threadIdx.x; i < 16 * Kpad; i += 256)
    Ws[i] = Wt[(size_t)(n0 + i / Kpad) * Kpad + (i % Kpad)];
  __syncthreads();

  int wv = threadIdx.x >> 5, lane = threadIdx.x & 31;
  int hi = lane >> 4, cl = lane & 15;
  int m0 = blockIdx.x * 128 + wv * 16;

  v8f acc = vzero();
  const __bf16* Arow = A + (size_t)m0 * lda;
  for (int kc = 0; kc < Kpad; kc += 32) {
    if (kc + 32 < Kpad) __builtin_prefetch(Arow + kc + 32, 0, 1);
    v16bf a = load_a16(Arow + kc, lda, lane);
    v16bf b = load_b16(Ws + kc, Kpad, lane);
    acc = wmma_bf16(a, b, acc);
  }

  int col = n0 + cl;
  float bv = (col < Ncols) ? bias[col] : 0.f;
#pragma unroll
  for (int e = 0; e < 8; ++e) {
    int m = m0 + e + 8 * hi;
    float v = acc[e] + bv;
    if constexpr (EPI == 0) {
      if (col < DIMC) v *= qscale;
      ((__bf16*)Cout)[(size_t)m * ldc + col] = (__bf16)v;
    } else if constexpr (EPI == 1 || EPI == 3) {
      if (col < Ncols)
        ((float*)Cout)[(size_t)m * ldc + col] = v + resid[(size_t)m * ldc + col];
    } else {
      float gv = 0.5f * v * (1.f + erff(v * 0.70710678118654752f));
      ((__bf16*)Cout)[(size_t)m * ldc + col] = (__bf16)gv;
    }
  }
}

// ---------------- fused shifted-window attention (one (window,head) per block) ---
// qkvb: bf16 [32768][544] (q pre-scaled). posT: fp32 [961][6]. attnb: bf16 [32768][180]
__global__ __launch_bounds__(256) void attn_kernel(const __bf16* __restrict__ qkvb,
                                                   const float* __restrict__ posT,
                                                   __bf16* __restrict__ attnb) {
  extern __shared__ char smem[];
  __bf16* Qs = (__bf16*)smem;           // [256][32]
  __bf16* Ks = Qs + 256 * 32;           // [256][32]
  __bf16* Vt = Ks + 256 * 32;           // [32][256]  (V transposed)
  __bf16* Ps = Vt + 256 * 32;           // [8 waves][16][256]
  signed char* lab = (signed char*)(Ps + 8 * 16 * 256);

  int head = blockIdx.x % HEADS;
  int win  = blockIdx.x / HEADS;
  int bimg = win >> 4;
  int w16  = win & 15;
  int wy = w16 >> 2, wx = w16 & 3;

  // ---- stage Q,K,V^T for this (window,head); fold the cyclic shift into the gather
  {
    int t = threadIdx.x;                       // window token
    int y = t >> 4, x = t & 15;
    int sy = wy * WIN + y, sx = wx * WIN + x;  // shifted-image coords
    int oy = (sy + 8) & 63, ox = (sx + 8) & 63;
    size_t base = ((size_t)bimg * LTOT + (size_t)oy * 64 + ox) * 544;
    const __bf16* qrow = qkvb + base + head * HD;
#pragma unroll
    for (int c = 0; c < 32; ++c) {
      __bf16 qv = (c < HD) ? qrow[c]            : (__bf16)0.f;
      __bf16 kv = (c < HD) ? qrow[DIMC + c]     : (__bf16)0.f;
      __bf16 vv = (c < HD) ? qrow[2 * DIMC + c] : (__bf16)0.f;
      Qs[t * 32 + c] = qv;
      Ks[t * 32 + c] = kv;
      Vt[c * 256 + t] = vv;
    }
    int rowg = (sy < 48) ? 0 : ((sy < 56) ? 1 : 2);
    int colg = (sx < 48) ? 0 : ((sx < 56) ? 1 : 2);
    lab[t] = (signed char)(rowg * 3 + colg);
  }
  __syncthreads();

  int wv = threadIdx.x >> 5;
  int lane = threadIdx.x & 31;
  int hi = lane >> 4, cl = lane & 15;
  __bf16* Pw = Ps + wv * (16 * 256);

  for (int rt = 0; rt < 2; ++rt) {
    int rows0 = wv * 32 + rt * 16;
    // ---- S = (scaled Q) @ K^T : 16x256 strip, 16 WMMAs
    v16bf aq = load_a16(Qs + rows0 * 32, 32, lane);
    v8f acc[16];
#pragma unroll
    for (int ct = 0; ct < 16; ++ct) {
      v16bf bk = load_b16(Ks + ct * 16 * 32, 32, lane);
      acc[ct] = wmma_bf16(aq, bk, vzero());
    }
    // ---- + dynamic relative position bias + shift mask
#pragma unroll
    for (int ct = 0; ct < 16; ++ct) {
#pragma unroll
      for (int e = 0; e < 8; ++e) {
        int it = rows0 + e + 8 * hi;
        int jt = ct * 16 + cl;
        int idx = ((it >> 4) - ct + 15) * 31 + ((it & 15) - cl + 15);
        float bias = posT[idx * 6 + head];
        float mval = (lab[it] != lab[jt]) ? -100.f : 0.f;
        acc[ct][e] += bias + mval;
      }
    }
    // ---- row softmax (row lives in 16 lanes of one half-wave, across 16 tiles)
#pragma unroll
    for (int e = 0; e < 8; ++e) {
      float m = -3.4e38f;
#pragma unroll
      for (int ct = 0; ct < 16; ++ct) m = fmaxf(m, acc[ct][e]);
#pragma unroll
      for (int d = 1; d < 16; d <<= 1) m = fmaxf(m, __shfl_xor(m, d, 16));
      float s = 0.f;
#pragma unroll
      for (int ct = 0; ct < 16; ++ct) {
        float ev = __expf(acc[ct][e] - m);
        acc[ct][e] = ev; s += ev;
      }
#pragma unroll
      for (int d = 1; d < 16; d <<= 1) s += __shfl_xor(s, d, 16);
      float inv = 1.f / s;
      int lr = e + 8 * hi;
#pragma unroll
      for (int ct = 0; ct < 16; ++ct)
        Pw[lr * 256 + ct * 16 + cl] = (__bf16)(acc[ct][e] * inv);
    }
    // ---- O = P @ V : 2 col-tiles x 8 K-chunks of WMMA
#pragma unroll
    for (int nt = 0; nt < 2; ++nt) {
      v8f o = vzero();
#pragma unroll
      for (int kc = 0; kc < 8; ++kc) {
        v16bf ap = load_a16(Pw + kc * 32, 256, lane);
        v16bf bv = load_b16(Vt + nt * 16 * 256 + kc * 32, 256, lane);
        o = wmma_bf16(ap, bv, o);
      }
      int col = nt * 16 + cl;
      if (col < HD) {
        int ch = head * HD + col;
#pragma unroll
        for (int e = 0; e < 8; ++e) {
          int tt = rows0 + e + 8 * hi;
          int sy = wy * WIN + (tt >> 4), sx = wx * WIN + (tt & 15);
          int oy = (sy + 8) & 63, ox = (sx + 8) & 63; // un-shift
          attnb[((size_t)bimg * LTOT + (size_t)oy * 64 + ox) * DIMC + ch] = (__bf16)o[e];
        }
      }
    }
  }
}

// ---------------- depthwise 3x3 conv on V + per-channel spatial sum -------------
__global__ __launch_bounds__(256) void dwconv_kernel(const __bf16* __restrict__ qkvb,
                                                     const float* __restrict__ dww,
                                                     const float* __restrict__ dwb,
                                                     __bf16* __restrict__ convb,
                                                     float* __restrict__ convsum) {
  int bc = blockIdx.x;
  int bimg = bc / DIMC, c = bc % DIMC;
  float w[9];
#pragma unroll
  for (int i = 0; i < 9; ++i) w[i] = dww[c * 9 + i];
  float bb = dwb[c];
  float lsum = 0.f;
  for (int p = threadIdx.x; p < LTOT; p += 256) {
    int y = p >> 6, x = p & 63;
    float acc = bb;
#pragma unroll
    for (int dy = -1; dy <= 1; ++dy)
#pragma unroll
      for (int dx = -1; dx <= 1; ++dx) {
        int yy = y + dy, xx = x + dx;
        if (yy >= 0 && yy < 64 && xx >= 0 && xx < 64)
          acc += w[(dy + 1) * 3 + (dx + 1)] *
                 (float)qkvb[((size_t)bimg * LTOT + yy * 64 + xx) * 544 + 2 * DIMC + c];
      }
    convb[((size_t)bimg * LTOT + p) * DIMC + c] = (__bf16)acc;
    lsum += acc;
  }
  __shared__ float red[256];
  red[threadIdx.x] = lsum;
  __syncthreads();
  for (int s2 = 128; s2 > 0; s2 >>= 1) {
    if ((int)threadIdx.x < s2) red[threadIdx.x] += red[threadIdx.x + s2];
    __syncthreads();
  }
  if (threadIdx.x == 0) convsum[bc] = red[0];
}

// ---------------- channel attention (tiny) ----------------
__global__ void ca_kernel(const float* __restrict__ convsum,
                          const float* __restrict__ ca1w, const float* __restrict__ ca1b,
                          const float* __restrict__ ca2w, const float* __restrict__ ca2b,
                          float* __restrict__ ca) {
  int bimg = blockIdx.x;
  __shared__ float ml[DIMC];
  __shared__ float s1[6];
  int t = threadIdx.x;
  if (t < DIMC) ml[t] = convsum[bimg * DIMC + t] * (1.f / 4096.f);
  __syncthreads();
  if (t < 6) {
    float a = ca1b[t];
    for (int c = 0; c < DIMC; ++c) a += ml[c] * ca1w[t * DIMC + c];
    s1[t] = fmaxf(a, 0.f);
  }
  __syncthreads();
  if (t < DIMC) {
    float a = ca2b[t];
#pragma unroll
    for (int j = 0; j < 6; ++j) a += s1[j] * ca2w[t * 6 + j];
    ca[bimg * DIMC + t] = 1.f / (1.f + __expf(-a));
  }
}

// ---------------- combine attn + conv*ca -> bf16 [M][192] padded -----------------
__global__ void combine_kernel(const __bf16* __restrict__ attnb, const __bf16* __restrict__ convb,
                               const float* __restrict__ ca, __bf16* __restrict__ asum) {
  size_t i = (size_t)blockIdx.x * 256 + threadIdx.x;
  if (i >= (size_t)MROWS * 192) return;
  int m = (int)(i / 192), c = (int)(i % 192);
  float v = 0.f;
  if (c < DIMC) {
    int bimg = m >> 12;
    v = (float)attnb[(size_t)m * DIMC + c] +
        (float)convb[(size_t)m * DIMC + c] * ca[bimg * DIMC + c];
  }
  asum[i] = (__bf16)v;
}

// ================================================================================
extern "C" void kernel_launch(void* const* d_in, const int* in_sizes, int n_in,
                              void* d_out, int out_size, void* d_ws, size_t ws_size,
                              hipStream_t stream) {
  (void)in_sizes; (void)n_in; (void)out_size; (void)ws_size;
  const float* x        = (const float*)d_in[0];
  const float* norm1_g  = (const float*)d_in[1];
  const float* norm1_b  = (const float*)d_in[2];
  const float* qkv_w    = (const float*)d_in[3];
  const float* qkv_b    = (const float*)d_in[4];
  const float* dw_w     = (const float*)d_in[5];
  const float* dw_b     = (const float*)d_in[6];
  const float* ca1_w    = (const float*)d_in[7];
  const float* ca1_b    = (const float*)d_in[8];
  const float* ca2_w    = (const float*)d_in[9];
  const float* ca2_b    = (const float*)d_in[10];
  const float* pos_proj_w = (const float*)d_in[11];
  const float* pos_proj_b = (const float*)d_in[12];
  const float* p1g = (const float*)d_in[13]; const float* p1b = (const float*)d_in[14];
  const float* p1w = (const float*)d_in[15]; const float* p1bb = (const float*)d_in[16];
  const float* p2g = (const float*)d_in[17]; const float* p2b = (const float*)d_in[18];
  const float* p2w = (const float*)d_in[19]; const float* p2bb = (const float*)d_in[20];
  const float* p3g = (const float*)d_in[21]; const float* p3b = (const float*)d_in[22];
  const float* p3w = (const float*)d_in[23]; const float* p3bb = (const float*)d_in[24];
  const float* proj_w   = (const float*)d_in[25];
  const float* proj_b   = (const float*)d_in[26];
  const float* norm2_g  = (const float*)d_in[27];
  const float* norm2_b  = (const float*)d_in[28];
  const float* fc1_w    = (const float*)d_in[29];
  const float* fc1_b    = (const float*)d_in[30];
  const float* fc2_w    = (const float*)d_in[31];
  const float* fc2_b    = (const float*)d_in[32];

  char* ws = (char*)d_ws;
  size_t off = 0;
  auto take = [&](size_t bytes) -> void* {
    void* p = ws + off;
    off += (bytes + 255) & ~(size_t)255;
    return p;
  };
  __bf16* xn      = (__bf16*)take((size_t)MROWS * 192 * 2);  // LN1 out, reused for LN2
  __bf16* qkvb    = (__bf16*)take((size_t)MROWS * 544 * 2);
  __bf16* attnb   = (__bf16*)take((size_t)MROWS * 180 * 2);
  __bf16* convb   = (__bf16*)take((size_t)MROWS * 180 * 2);
  __bf16* asum    = (__bf16*)take((size_t)MROWS * 192 * 2);
  __bf16* h1      = (__bf16*)take((size_t)MROWS * 720 * 2);
  float*  x1      = (float*) take((size_t)MROWS * 180 * 4);
  __bf16* qkv_wt  = (__bf16*)take((size_t)544 * 192 * 2);
  __bf16* proj_wt = (__bf16*)take((size_t)192 * 192 * 2);
  __bf16* fc1_wt  = (__bf16*)take((size_t)720 * 192 * 2);
  __bf16* fc2_wt  = (__bf16*)take((size_t)192 * 720 * 2);
  float*  posT    = (float*) take((size_t)961 * 6 * 4);
  float*  convsum = (float*) take(1440 * 4);
  float*  cab     = (float*) take(1440 * 4);

  // weights -> bf16, transposed [Npad][Kpad]
  cvt_wt_kernel<<<(544 * 192 + 255) / 256, 256, 0, stream>>>(qkv_w, qkv_wt, 180, 540, 192, 544);
  cvt_wt_kernel<<<(192 * 192 + 255) / 256, 256, 0, stream>>>(proj_w, proj_wt, 180, 180, 192, 192);
  cvt_wt_kernel<<<(720 * 192 + 255) / 256, 256, 0, stream>>>(fc1_w, fc1_wt, 180, 720, 192, 720);
  cvt_wt_kernel<<<(192 * 720 + 255) / 256, 256, 0, stream>>>(fc2_w, fc2_wt, 720, 180, 720, 192);

  posbias_kernel<<<4, 256, 0, stream>>>(pos_proj_w, pos_proj_b,
                                        p1g, p1b, p1w, p1bb,
                                        p2g, p2b, p2w, p2bb,
                                        p3g, p3b, p3w, p3bb, posT);

  ln_kernel<<<MROWS / 8, 256, 0, stream>>>(x, norm1_g, norm1_b, xn);

  // QKV GEMM (q columns pre-scaled by HD^-0.5)
  gemm_bf16_kernel<0><<<dim3(MROWS / 128, 34), 256, 0, stream>>>(
      xn, 192, qkv_wt, qkv_b, 540, 192, qkvb, 544, nullptr, SCALE_F);

  // fused shifted-window attention: 128 windows x 6 heads
  size_t attn_smem = (size_t)(256 * 32 * 3 + 8 * 16 * 256) * sizeof(__bf16) + 256;
  attn_kernel<<<128 * HEADS, 256, attn_smem, stream>>>(qkvb, posT, attnb);

  // conv branch
  dwconv_kernel<<<8 * DIMC, 256, 0, stream>>>(qkvb, dw_w, dw_b, convb, convsum);
  ca_kernel<<<8, 192, 0, stream>>>(convsum, ca1_w, ca1_b, ca2_w, ca2_b, cab);
  combine_kernel<<<(int)(((size_t)MROWS * 192 + 255) / 256), 256, 0, stream>>>(attnb, convb, cab, asum);

  // proj GEMM + residual -> x1 (fp32)
  gemm_bf16_kernel<1><<<dim3(MROWS / 128, 12), 256, 0, stream>>>(
      asum, 192, proj_wt, proj_b, 180, 192, x1, 180, x, 1.f);

  // MLP
  ln_kernel<<<MROWS / 8, 256, 0, stream>>>(x1, norm2_g, norm2_b, xn);
  gemm_bf16_kernel<2><<<dim3(MROWS / 128, 45), 256, 0, stream>>>(
      xn, 192, fc1_wt, fc1_b, 720, 192, h1, 720, nullptr, 1.f);
  gemm_bf16_kernel<3><<<dim3(MROWS / 128, 12), 256, 0, stream>>>(
      h1, 720, fc2_wt, fc2_b, 180, 720, (float*)d_out, 180, x1, 1.f);
}
